// Attention_25142738551091
// MI455X (gfx1250) — matrix operand order repaired and checked by reading, not compiled, not run on hardware
//
#include <hip/hip_runtime.h>
#include <hip/hip_bf16.h>

typedef _Float16 half_t;
typedef __attribute__((ext_vector_type(16))) _Float16 v16h;
typedef __attribute__((ext_vector_type(8)))  _Float16 v8h;
typedef __attribute__((ext_vector_type(4)))  _Float16 v4h;
typedef __attribute__((ext_vector_type(8)))  float    v8f;
typedef __attribute__((ext_vector_type(4), aligned(4))) float v4f_u;  // alignment-tolerant

#define BM 64
#define BN 64
#define BK 32
#define LDT 40   // padded LDS row (f16): 80B rows keep 16B alignment, dodge bank conflicts

// ---------------------------------------------------------------------------
// LDS staging helpers
// ---------------------------------------------------------------------------
__device__ __forceinline__ void stage_a(const float* __restrict__ A, half_t* dst,
                                        int m0, int k0, int M, int K, int tid)
{
  if (m0 + BM <= M && k0 + BK <= K) {
    // full tile: float4 load -> half4 LDS store
    for (int e = tid; e < BM * 8; e += 128) {
      int r = e >> 3, c4 = e & 7;
      v4f_u v = *(const v4f_u*)&A[(long)(m0 + r) * K + k0 + c4 * 4];
      *(v4h*)&dst[r * LDT + c4 * 4] = __builtin_convertvector(v, v4h);
    }
  } else {
    for (int e = tid; e < BM * BK; e += 128) {
      int r = e >> 5, c = e & 31;
      int gm = m0 + r, gk = k0 + c;
      float v = (gm < M && gk < K) ? A[(long)gm * K + gk] : 0.f;
      dst[r * LDT + c] = (half_t)v;
    }
  }
}

// B weights pre-converted to f16, [N][K] row-major. Async copy straight into LDS
// (CDNA5 GLOBAL_LOAD_ASYNC_TO_LDS_B128, ASYNCcnt) when tile is full & 16B-aligned.
__device__ __forceinline__ void stage_b_nk(const half_t* __restrict__ Bh, half_t* dst,
                                           int n0, int k0, int N, int K, int tid)
{
  if (n0 + BN <= N && k0 + BK <= K && (K & 7) == 0) {
    unsigned lbase = (unsigned)(size_t)&dst[0];                 // low 32b of flat = LDS offset
    unsigned long long sbase = (unsigned long long)(const void*)(Bh + (long)n0 * K + k0);
    for (int c = tid; c < 256; c += 128) {                      // 256 x 16B chunks
      int n = c >> 2, q = c & 3;
      unsigned laddr = lbase + (unsigned)(n * (LDT * 2) + q * 16);
      unsigned voff  = (unsigned)(n * K + q * 8) * 2u;          // byte offset from sbase
      asm volatile("global_load_async_to_lds_b128 %0, %1, %2"
                   :: "v"(laddr), "v"(voff), "s"(sbase) : "memory");
    }
  } else {
    for (int e = tid; e < BN * BK; e += 128) {
      int n = e >> 5, c = e & 31;
      int gn = n0 + n, gk = k0 + c;
      half_t v = (half_t)0.f;
      if (gn < N && gk < K) v = Bh[(long)gn * K + gk];
      dst[n * LDT + c] = v;
    }
  }
}

// B fp32 [K][N] row-major (x @ W weights): transposed store into LDS
__device__ __forceinline__ void stage_b_kn(const float* __restrict__ B, half_t* dst,
                                           int n0, int k0, int N, int K, int tid)
{
  if (n0 + BN <= N && k0 + BK <= K && (N & 3) == 0) {
    for (int e = tid; e < 512; e += 128) {
      int c = e >> 4, n4 = e & 15;
      v4f_u v = *(const v4f_u*)&B[(long)(k0 + c) * N + n0 + n4 * 4];
#pragma unroll
      for (int j = 0; j < 4; ++j) dst[(n4 * 4 + j) * LDT + c] = (half_t)v[j];
    }
  } else {
    for (int e = tid; e < BN * BK; e += 128) {
      int n = e & 63, c = e >> 6;
      int gn = n0 + n, gk = k0 + c;
      float v = (gn < N && gk < K) ? B[(long)gk * N + gn] : 0.f;
      dst[n * LDT + c] = (half_t)v;
    }
  }
}

// ---------------------------------------------------------------------------
// WMMA compute + epilogue shared by both GEMM flavors
// ---------------------------------------------------------------------------
__device__ __forceinline__ void gemm_core_step(const half_t* As, const half_t* Bs,
                                               int wm, int wn, int lane, v8f acc[2][2])
{
  const int lhalf = lane & 15;
  const int ksel  = (lane >> 4) * 8;    // A frag: runs {ksel..+7, 16+ksel..+7}
  const int kb    = (lane >> 4) * 8;    // B frag: runs {kb..+7, 16+kb..+7}
  v16h af[2], bf[2];
#pragma unroll
  for (int mi = 0; mi < 2; ++mi) {
    const half_t* p = &As[(wm + mi * 16 + lhalf) * LDT + ksel];
    v8h lo = *(const v8h*)p;
    v8h hi = *(const v8h*)(p + 16);
    af[mi] = __builtin_shufflevector(lo, hi, 0,1,2,3,4,5,6,7,8,9,10,11,12,13,14,15);
  }
#pragma unroll
  for (int ni = 0; ni < 2; ++ni) {
    const half_t* p = &Bs[(wn + ni * 16 + lhalf) * LDT + kb];
    v8h lo = *(const v8h*)p;
    v8h hi = *(const v8h*)(p + 16);
    bf[ni] = __builtin_shufflevector(lo, hi, 0,1,2,3,4,5,6,7,8,9,10,11,12,13,14,15);
  }
#pragma unroll
  for (int mi = 0; mi < 2; ++mi)
#pragma unroll
    for (int ni = 0; ni < 2; ++ni)
      acc[mi][ni] = __builtin_amdgcn_wmma_f32_16x16x32_f16(
          false, af[mi], false, bf[ni], (short)0, acc[mi][ni], false, false);
}

__device__ __forceinline__ void gemm_epilogue(v8f acc[2][2], float* __restrict__ C,
                                              const float* __restrict__ bias,
                                              const float* __restrict__ mul,
                                              int m0, int n0, int wm, int wn, int lane,
                                              int M, int N, int mul_ld, int mul_ofs)
{
#pragma unroll
  for (int mi = 0; mi < 2; ++mi)
#pragma unroll
    for (int ni = 0; ni < 2; ++ni) {
      int tm = m0 + wm + mi * 16 + (lane >> 4) * 8;
      int tn = n0 + wn + ni * 16 + (lane & 15);
      if (tn >= N) continue;
      float bv = bias ? bias[tn] : 0.f;
#pragma unroll
      for (int r = 0; r < 8; ++r) {
        int gm = tm + r;
        if (gm < M) {
          float v = acc[mi][ni][r] + bv;
          if (mul) v *= mul[(long)gm * mul_ld + mul_ofs + tn];
          C[(long)gm * N + tn] = v;
        }
      }
    }
}

// ---------------------------------------------------------------------------
// GEMM, weights f16 [N][K] (1x1-conv layout), async B staging, double-buffered
// ---------------------------------------------------------------------------
__global__ __launch_bounds__(128) void k_gemm_nk(
    const float* __restrict__ A, const half_t* __restrict__ Bh,
    const float* __restrict__ bias, const float* __restrict__ mul,
    float* __restrict__ C, int M, int N, int K, int mul_ld, int mul_ofs)
{
  __shared__ __align__(16) half_t As[2][BM * LDT];
  __shared__ __align__(16) half_t Bs[2][BN * LDT];
  const int tid = threadIdx.x, wave = tid >> 5, lane = tid & 31;
  const int m0 = blockIdx.x * BM, n0 = blockIdx.y * BN;
  const int wm = (wave >> 1) * 32, wn = (wave & 1) * 32;
  v8f acc[2][2] = {};
  const int nsteps = (K + BK - 1) / BK;

  stage_a(A, As[0], m0, 0, M, K, tid);
  stage_b_nk(Bh, Bs[0], n0, 0, N, K, tid);
  asm volatile("s_wait_asynccnt 0x0" ::: "memory");
  __syncthreads();

  for (int s = 0; s < nsteps; ++s) {
    const int buf = s & 1;
    if (s + 1 < nsteps) {
      stage_a(A, As[buf ^ 1], m0, (s + 1) * BK, M, K, tid);
      stage_b_nk(Bh, Bs[buf ^ 1], n0, (s + 1) * BK, N, K, tid);
    }
    gemm_core_step(As[buf], Bs[buf], wm, wn, lane, acc);
    asm volatile("s_wait_asynccnt 0x0" ::: "memory");
    __syncthreads();
  }
  gemm_epilogue(acc, C, bias, mul, m0, n0, wm, wn, lane, M, N, mul_ld, mul_ofs);
}

// ---------------------------------------------------------------------------
// GEMM, weights fp32 [K][N] (x @ W layout), double-buffered
// ---------------------------------------------------------------------------
__global__ __launch_bounds__(128) void k_gemm_kn(
    const float* __restrict__ A, const float* __restrict__ B,
    const float* __restrict__ bias, float* __restrict__ C, int M, int N, int K)
{
  __shared__ __align__(16) half_t As[2][BM * LDT];
  __shared__ __align__(16) half_t Bs[2][BN * LDT];
  const int tid = threadIdx.x, wave = tid >> 5, lane = tid & 31;
  const int m0 = blockIdx.x * BM, n0 = blockIdx.y * BN;
  const int wm = (wave >> 1) * 32, wn = (wave & 1) * 32;
  v8f acc[2][2] = {};
  const int nsteps = (K + BK - 1) / BK;

  stage_a(A, As[0], m0, 0, M, K, tid);
  stage_b_kn(B, Bs[0], n0, 0, N, K, tid);
  __syncthreads();

  for (int s = 0; s < nsteps; ++s) {
    const int buf = s & 1;
    if (s + 1 < nsteps) {
      stage_a(A, As[buf ^ 1], m0, (s + 1) * BK, M, K, tid);
      stage_b_kn(B, Bs[buf ^ 1], n0, (s + 1) * BK, N, K, tid);
    }
    gemm_core_step(As[buf], Bs[buf], wm, wn, lane, acc);
    __syncthreads();
  }
  gemm_epilogue(acc, C, bias, nullptr, m0, n0, wm, wn, lane, M, N, 0, 0);
}

// ---------------------------------------------------------------------------
// small prep / data-movement / pointwise kernels
// ---------------------------------------------------------------------------
__global__ void k_f32_to_f16(const float* __restrict__ s, half_t* __restrict__ d, long n)
{
  long t = (long)blockIdx.x * blockDim.x + threadIdx.x;
  if (t < n) d[t] = (half_t)s[t];
}

// concat the two 3x3 pointwise weights [392][72]+[392][72] -> f16 [392][144], sum biases
__global__ void k_wcat(const float* __restrict__ pw, const float* __restrict__ pw1,
                       const float* __restrict__ pb, const float* __restrict__ pb1,
                       half_t* __restrict__ wcat, float* __restrict__ bcat)
{
  int t = blockIdx.x * blockDim.x + threadIdx.x;
  if (t < 392) bcat[t] = pb[t] + pb1[t];
  if (t >= 392 * 144) return;
  int oc = t / 144, j = t % 144;
  wcat[t] = (half_t)((j < 72) ? pw[oc * 72 + j] : pw1[oc * 72 + (j - 72)]);
}

// qkv (pixel-major [B*4096][192]) -> f channel-major [16 imgs][24 ch][4096]
__global__ void k_qkv_to_fcm(const float* __restrict__ qkv, float* __restrict__ f, int total)
{
  int t = blockIdx.x * blockDim.x + threadIdx.x;
  if (t >= total) return;
  int ch = t % 192;
  int p  = (t / 192) & 4095;
  int b  = t / (192 * 4096);
  f[((long)(b * 8 + ch / 24) * 24 + (ch % 24)) * 4096 + p] = qkv[t];
}

// generic depthwise conv, stride-parameterized (channel-major or pixel-major)
__global__ void k_dwconv(const float* __restrict__ in, const float* __restrict__ w,
                         const float* __restrict__ bias, float* __restrict__ out,
                         int nimg, int Cc, int ksz, int relu_in,
                         long ii_s, long ic_s, long ip_s,
                         long oi_s, long oc_s, long op_s, int cfast)
{
  long t = (long)blockIdx.x * blockDim.x + threadIdx.x;
  long total = (long)nimg * Cc * 4096;
  if (t >= total) return;
  int img, c, p;
  if (cfast) { c = (int)(t % Cc); p = (int)((t / Cc) & 4095); img = (int)(t / ((long)Cc * 4096)); }
  else       { p = (int)(t & 4095); c = (int)((t >> 12) % Cc); img = (int)(t / ((long)Cc * 4096)); }
  int y = p >> 6, x = p & 63;
  int R = ksz >> 1;
  float accv = bias[c];
  const float* ib = in + (long)img * ii_s + (long)c * ic_s;
  const float* wc = w + (long)c * ksz * ksz;
  for (int dy = -R; dy <= R; ++dy) {
    int yy = y + dy; if (yy < 0 || yy > 63) continue;
    for (int dx = -R; dx <= R; ++dx) {
      int xx = x + dx; if (xx < 0 || xx > 63) continue;
      float v = ib[(long)(yy * 64 + xx) * ip_s];
      if (relu_in) v = v > 0.f ? v : 0.f;
      accv += v * wc[(dy + R) * ksz + (dx + R)];
    }
  }
  out[(long)img * oi_s + (long)c * oc_s + (long)p * op_s] = accv;
}

// im2col for the two 3x3 pointwise branches: col[65536][144]
__global__ void k_im2col(const float* __restrict__ t0, const float* __restrict__ t1,
                         float* __restrict__ col, long total)
{
  long t = (long)blockIdx.x * blockDim.x + threadIdx.x;
  if (t >= total) return;
  int j = (int)(t % 144);
  long pixg = t / 144;
  int p = (int)(pixg & 4095), img = (int)(pixg >> 12);
  int y = p >> 6, x = p & 63;
  const float* src = (j < 72) ? t0 : t1;
  int jj = j % 72;
  int ic = jj / 9, tap = jj % 9;
  int yy = y + tap / 3 - 1, xx = x + tap % 3 - 1;
  float v = 0.f;
  if (yy >= 0 && yy < 64 && xx >= 0 && xx < 64)
    v = src[((long)img * 8 + ic) * 4096 + yy * 64 + xx];
  col[t] = v;
}

// attn1 = softmax_kk( sum_hd q[hd]*k[hd,kk] + rpb[kk]*sum_hd q[hd] )
__global__ void k_attn(const float* __restrict__ fcm, const float* __restrict__ kexp,
                       const float* __restrict__ rpb, float* __restrict__ attn)
{
  int t = blockIdx.x * blockDim.x + threadIdx.x;
  if (t >= 16 * 4096) return;
  int p = t & 4095, image = t >> 12;
  int b = image >> 3, nh = image & 7;
  float q[8]; float qsum = 0.f;
  const float* fb = fcm + (long)image * 24 * 4096 + p;
#pragma unroll
  for (int hd = 0; hd < 8; ++hd) { q[hd] = fb[hd * 4096]; qsum += q[hd]; }
  const float* kbp = kexp + (long)t * 392;
  const float* rp  = rpb + nh * 49;
  float lg[49]; float mx = -1e30f;
  for (int kk = 0; kk < 49; ++kk) {
    float d = rp[kk] * qsum;
#pragma unroll
    for (int hd = 0; hd < 8; ++hd) d += q[hd] * kbp[hd * 49 + kk];
    lg[kk] = d; mx = fmaxf(mx, d);
  }
  float s = 0.f;
  for (int kk = 0; kk < 49; ++kk) { float e = __expf(lg[kk] - mx); lg[kk] = e; s += e; }
  float inv = 1.f / s;
  float* ab = attn + ((long)b * 4096 + p) * 392 + nh * 49;
  for (int kk = 0; kk < 49; ++kk) ab[kk] = lg[kk] * inv;
}

// a0 = fused[:, :98] * g[:, :98]
__global__ void k_mul_slice(const float* __restrict__ fused, const float* __restrict__ g,
                            float* __restrict__ a0, long total)
{
  long t = (long)blockIdx.x * blockDim.x + threadIdx.x;
  if (t >= total) return;
  int c = (int)(t % 98); long pix = t / 98;
  a0[t] = fused[pix * 784 + c] * g[pix * 686 + c];
}

// xo[b,p,nh*8+hd] = sum_kk (apout + attn)[b,p,nh*49+kk] * v[img,p,hd*49+kk]
__global__ void k_combine(const float* __restrict__ apout, const float* __restrict__ attn,
                          const float* __restrict__ vexp, float* __restrict__ xo)
{
  int t = blockIdx.x * blockDim.x + threadIdx.x;
  if (t >= 8192 * 64) return;
  int c = t & 63; long pix = t >> 6;
  int nh = c >> 3, hd = c & 7;
  long b = pix >> 12, p = pix & 4095;
  long image = b * 8 + nh;
  const float* ap = apout + pix * 392 + nh * 49;
  const float* at = attn  + pix * 392 + nh * 49;
  const float* vb = vexp  + (image * 4096 + p) * 392 + hd * 49;
  float acc = 0.f;
  for (int kk = 0; kk < 49; ++kk) acc += (ap[kk] + at[kk]) * vb[kk];
  xo[t] = acc;
}

// ---------------------------------------------------------------------------
extern "C" void kernel_launch(void* const* d_in, const int* in_sizes, int n_in,
                              void* d_out, int out_size, void* d_ws, size_t ws_size,
                              hipStream_t stream)
{
  const float* x        = (const float*)d_in[0];
  const float* w_qkv    = (const float*)d_in[1];
  const float* b_qkv    = (const float*)d_in[2];
  const float* dc_dw_w  = (const float*)d_in[3];
  const float* dc_dw_b  = (const float*)d_in[4];
  const float* dc_pw_w  = (const float*)d_in[5];
  const float* dc_pw_b  = (const float*)d_in[6];
  const float* dc1_dw_w = (const float*)d_in[7];
  const float* dc1_dw_b = (const float*)d_in[8];
  const float* dc1_pw_w = (const float*)d_in[9];
  const float* dc1_pw_b = (const float*)d_in[10];
  const float* rpb      = (const float*)d_in[11];
  const float* pin_w    = (const float*)d_in[12];
  const float* pin_b    = (const float*)d_in[13];
  const float* dw7_w    = (const float*)d_in[14];
  const float* dw7_b    = (const float*)d_in[15];
  const float* dw5_w    = (const float*)d_in[16];
  const float* dw5_b    = (const float*)d_in[17];
  const float* dw3_w    = (const float*)d_in[18];
  const float* dw3_b    = (const float*)d_in[19];
  const float* pout_w   = (const float*)d_in[20];
  const float* pout_b   = (const float*)d_in[21];
  const float* pws0_w   = (const float*)d_in[22];
  const float* pws0_b   = (const float*)d_in[23];
  const float* pws1_w   = (const float*)d_in[24];
  const float* pws1_b   = (const float*)d_in[25];
  const float* proj_w   = (const float*)d_in[26];
  const float* proj_b   = (const float*)d_in[27];

  float* ws = (float*)d_ws;
  size_t o = 0;
  auto alloc = [&](size_t n) { float* p = ws + o; o += n; return p; };
  float*  qkv_pm = alloc(8192L * 192);
  float*  f_cm   = alloc(16L * 24 * 4096);
  float*  t0     = alloc(16L * 8 * 4096);
  float*  t1     = alloc(16L * 8 * 4096);
  float*  bcat   = alloc(392);
  half_t* wcat_h = (half_t*)alloc(392L * 144 / 2);
  half_t* pin_h  = (half_t*)alloc(784L * 392 / 2);
  half_t* pws0_h = (half_t*)alloc(196L * 98 / 2 + 1);
  half_t* pws1_h = (half_t*)alloc(392L * 196 / 2);
  half_t* pout_h = (half_t*)alloc(392L * 392 / 2);
  float*  col    = alloc(65536L * 144);
  float*  kexp   = alloc(65536L * 392);
  float*  vexp   = alloc(65536L * 392);
  float*  attn   = alloc(8192L * 392);
  float*  fused  = alloc(8192L * 784);
  float*  g0     = alloc(8192L * 686);
  float*  g1     = alloc(8192L * 686);
  float*  a0     = alloc(8192L * 98);
  float*  a1     = alloc(8192L * 196);
  float*  a2     = alloc(8192L * 392);
  float*  apout  = alloc(8192L * 392);
  float*  xo     = alloc(8192L * 64);

  auto gemm_nk = [&](const float* A, const half_t* Bh, const float* bias, const float* mul,
                     float* Cout, int M, int N, int K, int mul_ld, int mul_ofs) {
    dim3 grid((M + BM - 1) / BM, (N + BN - 1) / BN);
    k_gemm_nk<<<grid, 128, 0, stream>>>(A, Bh, bias, mul, Cout, M, N, K, mul_ld, mul_ofs);
  };
  auto gemm_kn = [&](const float* A, const float* B, const float* bias,
                     float* Cout, int M, int N, int K) {
    dim3 grid((M + BM - 1) / BM, (N + BN - 1) / BN);
    k_gemm_kn<<<grid, 128, 0, stream>>>(A, B, bias, Cout, M, N, K);
  };
  auto blk = [](long n) { return (unsigned)((n + 255) / 256); };

  // weight prep (f16 conversions + concat)
  k_wcat<<<blk(392 * 144), 256, 0, stream>>>(dc_pw_w, dc1_pw_w, dc_pw_b, dc1_pw_b, wcat_h, bcat);
  k_f32_to_f16<<<blk(784L * 392), 256, 0, stream>>>(pin_w,  pin_h,  784L * 392);
  k_f32_to_f16<<<blk(196L * 98),  256, 0, stream>>>(pws0_w, pws0_h, 196L * 98);
  k_f32_to_f16<<<blk(392L * 196), 256, 0, stream>>>(pws1_w, pws1_h, 392L * 196);
  k_f32_to_f16<<<blk(392L * 392), 256, 0, stream>>>(pout_w, pout_h, 392L * 392);

  // qkv projection: [8192,256] x [256,192]
  gemm_kn(x, w_qkv, b_qkv, qkv_pm, 8192, 192, 256);
  k_qkv_to_fcm<<<blk(2L * 4096 * 192), 256, 0, stream>>>(qkv_pm, f_cm, 2 * 4096 * 192);

  const long fimg = 24L * 4096, timg = 8L * 4096;
  // k branch: depthwise 7x7 (both branches) -> im2col -> GEMM (K=144, N=392)
  k_dwconv<<<blk(16L * 8 * 4096), 256, 0, stream>>>(f_cm + 8 * 4096, dc_dw_w, dc_dw_b, t0,
      16, 8, 7, 0, fimg, 4096, 1, timg, 4096, 1, 0);
  k_dwconv<<<blk(16L * 8 * 4096), 256, 0, stream>>>(f_cm + 8 * 4096, dc1_dw_w, dc1_dw_b, t1,
      16, 8, 7, 0, fimg, 4096, 1, timg, 4096, 1, 0);
  k_im2col<<<blk(65536L * 144), 256, 0, stream>>>(t0, t1, col, 65536L * 144);
  gemm_nk(col, wcat_h, bcat, nullptr, kexp, 65536, 392, 144, 0, 0);

  // v branch
  k_dwconv<<<blk(16L * 8 * 4096), 256, 0, stream>>>(f_cm + 16 * 4096, dc_dw_w, dc_dw_b, t0,
      16, 8, 7, 0, fimg, 4096, 1, timg, 4096, 1, 0);
  k_dwconv<<<blk(16L * 8 * 4096), 256, 0, stream>>>(f_cm + 16 * 4096, dc1_dw_w, dc1_dw_b, t1,
      16, 8, 7, 0, fimg, 4096, 1, timg, 4096, 1, 0);
  k_im2col<<<blk(65536L * 144), 256, 0, stream>>>(t0, t1, col, 65536L * 144);
  gemm_nk(col, wcat_h, bcat, nullptr, vexp, 65536, 392, 144, 0, 0);

  // softmax attention
  k_attn<<<blk(16L * 4096), 256, 0, stream>>>(f_cm, kexp, rpb, attn);

  // pin 1x1: 392 -> 784
  gemm_nk(attn, pin_h, pin_b, nullptr, fused, 8192, 784, 392, 0, 0);

  // gating chain: dw7 -> relu.dw5 -> relu.dw3 (pixel-major, channels-fastest)
  k_dwconv<<<blk(2L * 686 * 4096), 256, 0, stream>>>(fused + 98, dw7_w, dw7_b, g0,
      2, 686, 7, 0, 4096L * 784, 1, 784, 4096L * 686, 1, 686, 1);
  k_dwconv<<<blk(2L * 686 * 4096), 256, 0, stream>>>(g0, dw5_w, dw5_b, g1,
      2, 686, 5, 1, 4096L * 686, 1, 686, 4096L * 686, 1, 686, 1);
  k_dwconv<<<blk(2L * 686 * 4096), 256, 0, stream>>>(g1, dw3_w, dw3_b, g0,
      2, 686, 3, 1, 4096L * 686, 1, 686, 4096L * 686, 1, 686, 1);

  // gated pointwise ladder (gate fused into GEMM epilogue)
  k_mul_slice<<<blk(8192L * 98), 256, 0, stream>>>(fused, g0, a0, 8192L * 98);
  gemm_nk(a0, pws0_h, pws0_b, g0, a1, 8192, 196, 98, 686, 98);
  gemm_nk(a1, pws1_h, pws1_b, g0, a2, 8192, 392, 196, 686, 294);
  gemm_nk(a2, pout_h, pout_b, nullptr, apout, 8192, 392, 392, 0, 0);

  // (a + attn1) * v, reduce over KK
  k_combine<<<blk(8192L * 64), 256, 0, stream>>>(apout, attn, vexp, xo);

  // final projection -> d_out [8192][256]
  gemm_kn(xo, proj_w, proj_b, (float*)d_out, 8192, 256, 64);
}
